// MultiHeadAttention_31009663877897
// MI455X (gfx1250) — matrix operand order repaired and checked by reading, not compiled, not run on hardware
//
#include <hip/hip_runtime.h>
#include <stdint.h>

typedef __attribute__((ext_vector_type(16))) _Float16 v16h;
typedef __attribute__((ext_vector_type(8)))  _Float16 v8h;
typedef __attribute__((ext_vector_type(8)))  float    v8f;

// ---------------------------------------------------------------------------
// WMMA helper (gfx1250: v_wmma_f32_16x16x32_f16, D = A(16x32) * B(32x16) + C)
// ---------------------------------------------------------------------------
__device__ __forceinline__ v8f wmma_f16(v16h a, v16h b, v8f c) {
  return __builtin_amdgcn_wmma_f32_16x16x32_f16(
      /*neg_a=*/false, a, /*neg_b=*/false, b,
      /*c_mod=*/(short)0, c, /*reuse_a=*/false, /*reuse_b=*/false);
}

// Raw IEEE maximumNumber without the compiler's sNaN canonicalize ops
// (WMMA outputs are never sNaN).
__device__ __forceinline__ float fmax_raw(float a, float b) {
  float d;
  asm("v_max_num_f32 %0, %1, %2" : "=v"(d) : "v"(a), "v"(b));
  return d;
}

// A fragment 16x32 f16, row-major source with leading dim `ld` (elements).
__device__ __forceinline__ v16h load_a_frag(const _Float16* base, int ld) {
  int l  = threadIdx.x & 31;
  int m  = l & 15;
  int kh = (l >> 4) * 8;
  union { v16h v; v8h h[2]; } u;
  u.h[0] = *(const v8h*)(base + (size_t)m * ld + kh);
  u.h[1] = *(const v8h*)(base + (size_t)m * ld + 16 + kh);
  return u.v;
}

// B fragment 32x16 f16 from a COLUMN-MAJOR source (leading dim `ld`).
__device__ __forceinline__ v16h load_b_frag(const _Float16* colbase, int ld) {
  int l  = threadIdx.x & 31;
  int n  = l & 15;
  int kq = (l >> 4) * 8;
  const _Float16* p = colbase + (size_t)n * ld + kq;
  union { v16h v; v8h h[2]; } u;
  u.h[0] = *(const v8h*)(p);
  u.h[1] = *(const v8h*)(p + 16);
  return u.v;
}

// ---------------------------------------------------------------------------
// Pure-VALU 16-lane max reduction via DPP (stays within a 16-lane DPP row).
// ---------------------------------------------------------------------------
__device__ __forceinline__ float row16_max(float x) {
  int t;
  t = __builtin_amdgcn_update_dpp(0, __builtin_bit_cast(int, x), 0xB1, 0xf, 0xf, true);
  x = fmax_raw(x, __builtin_bit_cast(float, t));
  t = __builtin_amdgcn_update_dpp(0, __builtin_bit_cast(int, x), 0x4E, 0xf, 0xf, true);
  x = fmax_raw(x, __builtin_bit_cast(float, t));
  t = __builtin_amdgcn_update_dpp(0, __builtin_bit_cast(int, x), 0x141, 0xf, 0xf, true);
  x = fmax_raw(x, __builtin_bit_cast(float, t));
  t = __builtin_amdgcn_update_dpp(0, __builtin_bit_cast(int, x), 0x140, 0xf, 0xf, true);
  x = fmax_raw(x, __builtin_bit_cast(float, t));
  return x;
}

// ---------------------------------------------------------------------------
// CDNA5 async global -> LDS copy (16B/lane), tracked by ASYNCcnt.
// ---------------------------------------------------------------------------
__device__ __forceinline__ void async_copy_b128(void* lds_ptr, const void* gptr) {
  uint32_t loff  = (uint32_t)(uintptr_t)lds_ptr;
  uint64_t gaddr = (uint64_t)(uintptr_t)gptr;
  asm volatile("global_load_async_to_lds_b128 %0, %1, off"
               :: "v"(loff), "v"(gaddr) : "memory");
}
__device__ __forceinline__ void wait_async_le4() {
  asm volatile("s_wait_asynccnt 4" ::: "memory");
}

// ---------------------------------------------------------------------------
// Converters
// ---------------------------------------------------------------------------
__global__ void cvt_f32_f16(const float* __restrict__ in, _Float16* __restrict__ out,
                            size_t n) {
  size_t stride = (size_t)gridDim.x * blockDim.x;
  for (size_t i = (size_t)blockIdx.x * blockDim.x + threadIdx.x; i < n; i += stride)
    out[i] = (_Float16)in[i];
}

// Wt[n*K + k] = (f16) W[k*N + n]
__global__ void cvt_w_transpose(const float* __restrict__ W, _Float16* __restrict__ Wt,
                                int K, int N) {
  size_t total = (size_t)K * N;
  size_t stride = (size_t)gridDim.x * blockDim.x;
  for (size_t i = (size_t)blockIdx.x * blockDim.x + threadIdx.x; i < total; i += stride) {
    int k = (int)(i % K);
    int n = (int)(i / K);
    Wt[i] = (_Float16)W[(size_t)k * N + n];
  }
}

// ---------------------------------------------------------------------------
// Tiled GEMM with double-buffered async-to-LDS staging.
//   C[M,N] = (A[M,K] * Bt[N,K]^T + bias[N]) * outScale
// MODE 0: f16 row-major out.  MODE 1: f32 row-major out.
// MODE 2: f16 out written per-head transposed: Vt[((b*H+h)*64+d)*T + t]
// ---------------------------------------------------------------------------
template <int MODE>
__global__ __launch_bounds__(256)
void gemm128(const _Float16* __restrict__ A, const _Float16* __restrict__ Bt,
             const float* __restrict__ bias,
             _Float16* __restrict__ Ch, float* __restrict__ Cf,
             int M, int N, int K, float outScale, int Tdim, int Hheads) {
  constexpr int LDT = 40;  // padded leading dim (f16), rows 16B aligned
  __shared__ _Float16 sA[2][128 * LDT];
  __shared__ _Float16 sB[2][128 * LDT];

  const int tid  = threadIdx.x;
  const int wave = tid >> 5;
  const int lane = tid & 31;
  const int bm = blockIdx.x * 128;
  const int bn = blockIdx.y * 128;
  const int wm = (wave >> 1) * 32;  // 0,32,64,96
  const int wn = (wave & 1) * 64;   // 0,64

  v8f acc[2][4] = {};

  const int srow  = tid >> 1;
  const int shalf = (tid & 1) * 16;
  const _Float16* gA = A + (size_t)(bm + srow) * K + shalf;
  const _Float16* gB = Bt + (size_t)(bn + srow) * K + shalf;
  const int soff = srow * LDT + shalf;

  const int steps = K / 32;

  // Prologue: stage tile 0 into buffer 0.
  async_copy_b128(&sA[0][soff],     gA);
  async_copy_b128(&sA[0][soff + 8], gA + 8);
  async_copy_b128(&sB[0][soff],     gB);
  async_copy_b128(&sB[0][soff + 8], gB + 8);

  for (int n = 0; n < steps; ++n) {
    // Prefetch next tile into the other buffer (clamped on the last step so
    // the in-order `asynccnt <= 4` wait still covers the current tile).
    const int knext = (n + 1 < steps) ? (n + 1) * 32 : n * 32;
    const int nb = (n + 1) & 1;
    async_copy_b128(&sA[nb][soff],     gA + knext);
    async_copy_b128(&sA[nb][soff + 8], gA + knext + 8);
    async_copy_b128(&sB[nb][soff],     gB + knext);
    async_copy_b128(&sB[nb][soff + 8], gB + knext + 8);

    wait_async_le4();   // current tile's 4 copies complete (in-order)
    __syncthreads();

    const _Float16* cA = sA[n & 1];
    const _Float16* cB = sB[n & 1];
    v16h af[2], bf[4];
#pragma unroll
    for (int i = 0; i < 2; ++i) af[i] = load_a_frag(cA + (wm + i * 16) * LDT, LDT);
#pragma unroll
    for (int j = 0; j < 4; ++j) bf[j] = load_b_frag(cB + (wn + j * 16) * LDT, LDT);
#pragma unroll
    for (int i = 0; i < 2; ++i)
#pragma unroll
      for (int j = 0; j < 4; ++j)
        acc[i][j] = wmma_f16(af[i], bf[j], acc[i][j]);

    __syncthreads();  // all waves done reading before this buffer is re-staged
  }

  const int nl = lane & 15;
  const int r0 = (lane >> 4) * 8;
#pragma unroll
  for (int i = 0; i < 2; ++i)
#pragma unroll
    for (int j = 0; j < 4; ++j) {
      const int gn = bn + wn + j * 16 + nl;
      const float bv = bias[gn];
#pragma unroll
      for (int r = 0; r < 8; ++r) {
        const int gm = bm + wm + i * 16 + r0 + r;
        const float val = (acc[i][j][r] + bv) * outScale;
        if (MODE == 1) {
          Cf[(size_t)gm * N + gn] = val;
        } else if (MODE == 0) {
          Ch[(size_t)gm * N + gn] = (_Float16)val;
        } else {
          // per-head transposed V: gm = b*T + t ; gn = h*64 + d
          const int b = gm / Tdim, t = gm % Tdim;
          const int h = gn >> 6, d = gn & 63;
          Ch[(((size_t)(b * Hheads + h) * 64 + d) * Tdim) + t] = (_Float16)val;
        }
      }
    }
}

// ---------------------------------------------------------------------------
// Flash attention (causal). One wave = 16 query rows of one (b,h).
// Logits arrive pre-scaled by 0.125*log2(e) (folded into Q projection), so the
// whole online softmax runs in base-2 with bare v_exp_f32.
// Only the single diagonal tile applies the causal mask.
// ---------------------------------------------------------------------------
template <bool MASKED>
__device__ __forceinline__ void fa_tile(
    int key0, int q0, int nl, int r0,
    v16h qf0, v16h qf1, v16h ones,
    const _Float16* kbase, const _Float16* vtbase, _Float16* myP,
    int C, int T, float* mrow, float* lrow, v8f* o) {
  // ---- S = Q * K^T : two 16x16 tiles over 32 keys ----
  v8f s0 = {}, s1 = {};
  v16h kb = load_b_frag(kbase + (size_t)key0 * C, C);
  s0 = wmma_f16(qf0, kb, s0);
  kb = load_b_frag(kbase + (size_t)key0 * C + 32, C);
  s0 = wmma_f16(qf1, kb, s0);
  kb = load_b_frag(kbase + (size_t)(key0 + 16) * C, C);
  s1 = wmma_f16(qf0, kb, s1);
  kb = load_b_frag(kbase + (size_t)(key0 + 16) * C + 32, C);
  s1 = wmma_f16(qf1, kb, s1);

  // ---- online softmax (rows in VGPRs, cols across 16 lanes), base-2 ----
  float alpha_s[8], p0[8], p1[8];
#pragma unroll
  for (int r = 0; r < 8; ++r) {
    float a0 = s0[r];
    float a1 = s1[r];
    if (MASKED) {
      const int q = q0 + r0 + r;
      if (key0 + nl > q)      a0 = -3.0e38f;
      if (key0 + 16 + nl > q) a1 = -3.0e38f;
    }
    const float mx = row16_max(fmax_raw(a0, a1));
    const float mnew = fmax_raw(mrow[r], mx);
    alpha_s[r] = __builtin_amdgcn_exp2f(mrow[r] - mnew);
    p0[r] = __builtin_amdgcn_exp2f(a0 - mnew);
    p1[r] = __builtin_amdgcn_exp2f(a1 - mnew);
    mrow[r] = mnew;
  }
#pragma unroll
  for (int dt = 0; dt < 4; ++dt)
#pragma unroll
    for (int r = 0; r < 8; ++r) o[dt][r] *= alpha_s[r];

  // ---- transpose P through per-wave LDS: C/D layout -> A fragment ----
#pragma unroll
  for (int r = 0; r < 8; ++r) {
    myP[(r0 + r) * 32 + nl]      = (_Float16)p0[r];
    myP[(r0 + r) * 32 + 16 + nl] = (_Float16)p1[r];
  }
  asm volatile("s_wait_dscnt 0" ::: "memory");  // same-wave LDS ops in-order
  const v16h pf = load_a_frag(myP, 32);

  // ---- row sums via WMMA against ones (broadcast across lanes) ----
  v8f zero = {};
  const v8f rs = wmma_f16(pf, ones, zero);
#pragma unroll
  for (int r = 0; r < 8; ++r) lrow[r] = lrow[r] * alpha_s[r] + rs[r];

  // ---- O += P * V ----
#pragma unroll
  for (int dt = 0; dt < 4; ++dt) {
    v16h vb = load_b_frag(vtbase + (size_t)(dt * 16) * T + key0, T);
    o[dt] = wmma_f16(pf, vb, o[dt]);
  }
}

__global__ __launch_bounds__(128)
void flash_attn(const _Float16* __restrict__ Qh, const _Float16* __restrict__ Kh,
                const _Float16* __restrict__ Vt, _Float16* __restrict__ Yh,
                int B, int H, int T, int C) {
  __shared__ _Float16 sP[4][16 * 32];  // per-wave P tile

  const int wave = threadIdx.x >> 5;
  const int lane = threadIdx.x & 31;
  const int widx = blockIdx.x * 4 + wave;
  const int qtPerSeq = T / 16;
  const int qb = widx % qtPerSeq;
  const int bh = widx / qtPerSeq;
  const int b = bh / H;
  const int h = bh % H;
  const int q0 = qb * 16;
  const int D = 64;

  const _Float16* qbase = Qh + ((size_t)(b * T + q0)) * C + h * D;
  const v16h qf0 = load_a_frag(qbase, C);      // scale*log2e already folded in
  const v16h qf1 = load_a_frag(qbase + 32, C);

  const _Float16* kbase  = Kh + ((size_t)(b * T)) * C + h * D;
  const _Float16* vtbase = Vt + ((size_t)bh * D) * T;
  _Float16* myP = sP[wave];

  v16h ones;
#pragma unroll
  for (int i = 0; i < 16; ++i) ones[i] = (_Float16)1.0f;

  float mrow[8], lrow[8];
  v8f o[4] = {};
#pragma unroll
  for (int r = 0; r < 8; ++r) { mrow[r] = -3.0e38f; lrow[r] = 0.0f; }

  const int nl = lane & 15;
  const int r0 = (lane >> 4) * 8;

  // Tiles fully below the diagonal need no mask; exactly one diagonal tile.
  const int nfull = (q0 + 1) / 32;
  for (int kt = 0; kt < nfull; ++kt)
    fa_tile<false>(kt * 32, q0, nl, r0, qf0, qf1, ones,
                   kbase, vtbase, myP, C, T, mrow, lrow, o);
  fa_tile<true>(nfull * 32, q0, nl, r0, qf0, qf1, ones,
                kbase, vtbase, myP, C, T, mrow, lrow, o);

  // ---- normalize (v_rcp_f32, adequate for f16 output) and write y ----
  float linv[8];
#pragma unroll
  for (int r = 0; r < 8; ++r) linv[r] = __builtin_amdgcn_rcpf(lrow[r]);
#pragma unroll
  for (int dt = 0; dt < 4; ++dt)
#pragma unroll
    for (int r = 0; r < 8; ++r) {
      const int q = q0 + r0 + r;
      const int d = dt * 16 + nl;
      Yh[((size_t)(b * T + q)) * C + h * D + d] = (_Float16)(o[dt][r] * linv[r]);
    }
}

// ---------------------------------------------------------------------------
// Host-side launch
// ---------------------------------------------------------------------------
extern "C" void kernel_launch(void* const* d_in, const int* in_sizes, int n_in,
                              void* d_out, int out_size, void* d_ws, size_t ws_size,
                              hipStream_t stream) {
  (void)in_sizes; (void)n_in; (void)out_size; (void)ws_size;

  const float* x  = (const float*)d_in[0];
  const float* Wq = (const float*)d_in[1];
  const float* bq = (const float*)d_in[2];
  const float* Wk = (const float*)d_in[3];
  const float* bk = (const float*)d_in[4];
  const float* Wv = (const float*)d_in[5];
  const float* bv = (const float*)d_in[6];
  const float* Wp = (const float*)d_in[7];
  const float* bp = (const float*)d_in[8];

  const int B = 4, T = 2048, C = 1024, H = 16;
  const size_t MT = (size_t)B * T;  // 8192 rows
  // 1/sqrt(64) * log2(e): softmax runs in base-2 downstream.
  const float attScale = 0.125f * 1.44269504088896340736f;

  _Float16* ws  = (_Float16*)d_ws;
  _Float16* x_h = ws;                                // MT*C
  _Float16* Wqt = x_h + MT * C;                      // C*C each
  _Float16* Wkt = Wqt + (size_t)C * C;
  _Float16* Wvt = Wkt + (size_t)C * C;
  _Float16* Wpt = Wvt + (size_t)C * C;
  _Float16* Qh  = Wpt + (size_t)C * C;               // MT*C each
  _Float16* Kh  = Qh + MT * C;
  _Float16* Vtb = Kh + MT * C;                       // per-head transposed V
  _Float16* Yh  = Vtb + MT * C;

  cvt_f32_f16<<<1024, 256, 0, stream>>>(x, x_h, MT * C);
  cvt_w_transpose<<<512, 256, 0, stream>>>(Wq, Wqt, C, C);
  cvt_w_transpose<<<512, 256, 0, stream>>>(Wk, Wkt, C, C);
  cvt_w_transpose<<<512, 256, 0, stream>>>(Wv, Wvt, C, C);
  cvt_w_transpose<<<512, 256, 0, stream>>>(Wp, Wpt, C, C);

  dim3 gg((unsigned)(MT / 128), (unsigned)(C / 128));
  gemm128<0><<<gg, 256, 0, stream>>>(x_h, Wqt, bq, Qh, nullptr, (int)MT, C, C,
                                     attScale, T, H);
  gemm128<0><<<gg, 256, 0, stream>>>(x_h, Wkt, bk, Kh, nullptr, (int)MT, C, C,
                                     1.0f, T, H);
  gemm128<2><<<gg, 256, 0, stream>>>(x_h, Wvt, bv, Vtb, nullptr, (int)MT, C, C,
                                     1.0f, T, H);

  const int nwaves = B * H * (T / 16);               // 8192 waves
  flash_attn<<<nwaves / 4, 128, 0, stream>>>(Qh, Kh, Vtb, Yh, B, H, T, C);

  gemm128<1><<<gg, 256, 0, stream>>>(Yh, Wpt, bp, nullptr, (float*)d_out, (int)MT, C, C,
                                     1.0f, T, H);
}
